// FixedPointSineModel_40269613367827
// MI455X (gfx1250) — matrix operand order repaired and checked by reading, not compiled, not run on hardware
//
#include <hip/hip_runtime.h>

typedef __attribute__((ext_vector_type(16))) _Float16 v16h;
typedef __attribute__((ext_vector_type(8)))  float    v8f;

#define B_SZ 2048
#define T_SZ 1024
#define H_SZ 64
#define G4   256

// Q8.8 fake-quant: clip(round(x*256)/256, -128, 127.99609375). rintf = RNE = jnp.round.
__device__ __forceinline__ float quantf(float x) {
  float q = rintf(x * 256.0f) * 0.00390625f;
  return fminf(fmaxf(q, -128.0f), 127.99609375f);
}
__device__ __forceinline__ float hsig(float x) {   // hardsigmoid
  return fminf(fmaxf(x * 0.16666667f + 0.5f, 0.0f), 1.0f);
}
__device__ __forceinline__ float htanh(float x) {  // hardtanh
  return fminf(fmaxf(x, -1.0f), 1.0f);
}

// Block = 128 threads = 4 waves = one 16-row batch tile.
// Wave ng computes gate columns {g*64 + ng*16 + [0,16)} for the tile's 16 rows.
// whh (quantized, f16, WMMA B layout) lives entirely in registers: 8 x v16h/wave.
// h state round-trips through a tiny double-buffered LDS tile in WMMA A layout.
__global__ __launch_bounds__(128)
void lstm_q88_wmma_kernel(const float* __restrict__ x,
                          const float* __restrict__ w_ih,
                          const float* __restrict__ w_hh,
                          const float* __restrict__ b_ih,
                          const float* __restrict__ b_hh,
                          const float* __restrict__ w_out,
                          const float* __restrict__ b_out,
                          float* __restrict__ out) {
  // h in 16-bit A-operand layout, double-buffered on t parity: [parity][kchunk][lane*16+e]
  __shared__ __attribute__((aligned(32))) _Float16 sHA[2][2][512];
  __shared__ float sX[2][16];   // quant(x[row,t]), double-buffered
  __shared__ float sOut[16];    // head reduction

  const int tid  = threadIdx.x;
  const int lane = tid & 31;
  const int ng   = tid >> 5;   // 16-column hidden-unit group (one per wave)
  const int nl   = lane & 15;
  const int rowBase = blockIdx.x * 16;

  // ---- one-time init ----
  // B layout (32x16 f16): lane = n + 16*(k>=16), e = k & 15  ->  for fixed lane the
  // 16 elements are 16 consecutive k: quantize+pack straight from global.
  v16h breg[4][2];
#pragma unroll
  for (int g = 0; g < 4; ++g) {
    const int n_global = g * 64 + ng * 16 + nl;
#pragma unroll
    for (int kc = 0; kc < 2; ++kc) {
      const int k_base = kc * 32 + ((lane & 16) ? 16 : 0);
      const float* wp = w_hh + n_global * 64 + k_base;
      v16h b;
#pragma unroll
      for (int e = 0; e < 16; ++e) b[e] = (_Float16)quantf(wp[e]);
      breg[g][kc] = b;
    }
  }
  // per-column input weight / bias (C/D layout: this lane owns col = g*64+ng*16+nl)
  float wihc[4], bc[4];
#pragma unroll
  for (int g = 0; g < 4; ++g) {
    int col = g * 64 + ng * 16 + nl;
    wihc[g] = quantf(w_ih[col]);
    bc[g]   = quantf(b_ih[col]) + quantf(b_hh[col]);
  }

  for (int idx = tid; idx < 2 * 2 * 512; idx += 128)
    ((_Float16*)sHA)[idx] = (_Float16)0.0f;           // h0 = 0 (both parities)
  if (ng == 0 && lane < 16)
    sX[0][lane] = quantf(x[(rowBase + lane) * T_SZ]); // stage x for t=0

  // A layout (16x32 f16, contiguous k as in B): lane = m + 16*k[4], e = k[3:0].
  // This wave's h elements live at hidden col kglob = ng*16 + nl.
  const int kglob  = ng * 16 + nl;
  const int kc_h   = kglob >> 5;
  const int klocal = kglob & 31;
  const int e_h    = klocal & 15;
  const int lhi    = klocal & 16;
  const int rowOffD = (lane & 16) ? 8 : 0; // C/D layout: lane half selects M or M+8

  v8f cst;                 // cell state c: one 16x16 f32 tile in registers
  float harr[8];           // latest h values (C/D layout)
#pragma unroll
  for (int r = 0; r < 8; ++r) { cst[r] = 0.0f; harr[r] = 0.0f; }

  __syncthreads();

  // ================= sequential scan over T (1 barrier / step) =================
  for (int t = 0; t < T_SZ; ++t) {
    const int p = t & 1;

    // A operands: h [16x64] as two 16x32 f16 chunks from LDS (parity p)
    const v16h* pA0 = (const v16h*)(&sHA[p][0][0]);
    const v16h* pA1 = (const v16h*)(&sHA[p][1][0]);
    v16h a0 = pA0[lane];
    v16h a1 = pA1[lane];

    float xv[8];
#pragma unroll
    for (int r = 0; r < 8; ++r) xv[r] = sX[p][r + rowOffD];

    // prefetch + stage x for t+1 into the other parity buffer
    if (ng == 0 && lane < 16 && t + 1 < T_SZ) {
      const float* xp = &x[(rowBase + lane) * T_SZ + t + 1];
      sX[p ^ 1][lane] = quantf(*xp);
      __builtin_prefetch(xp + 32, 0, 1);  // global_prefetch_b8: warm the x stream
    }

    v8f garr[4];
#pragma unroll
    for (int g = 0; g < 4; ++g) {
      v8f acc;
#pragma unroll
      for (int r = 0; r < 8; ++r) acc[r] = fmaf(xv[r], wihc[g], bc[g]); // xg_t seed
      acc = __builtin_amdgcn_wmma_f32_16x16x32_f16(false, a0, false, breg[g][0],
                                                   (short)0, acc, false, false);
      acc = __builtin_amdgcn_wmma_f32_16x16x32_f16(false, a1, false, breg[g][1],
                                                   (short)0, acc, false, false);
#pragma unroll
      for (int r = 0; r < 8; ++r) acc[r] = quantf(acc[r]); // g = quant(xg + h@whh.T)
      garr[g] = acc;
    }

    // LSTM cell update in C/D register layout; scatter new h to A layout (parity p^1)
#pragma unroll
    for (int r = 0; r < 8; ++r) {
      float ig = hsig(garr[0][r]);
      float fg = hsig(garr[1][r]);
      float gg = htanh(garr[2][r]);
      float og = hsig(garr[3][r]);
      float c  = quantf(fg * cst[r] + ig * gg);
      float h  = quantf(og * htanh(c));
      cst[r]  = c;
      harr[r] = h;
      int m = r + rowOffD;
      sHA[p ^ 1][kc_h][(m + lhi) * 16 + e_h] = (_Float16)h;
    }

    __syncthreads();   // publish h(t) / x(t+1) before next step's reads
  }

  // ================= linear head: out = quant(hT @ qwout + qbout) =================
  if (tid < 16) sOut[tid] = 0.0f;
  __syncthreads();
  {
    float wq = quantf(w_out[ng * 16 + nl]);
#pragma unroll
    for (int r = 0; r < 8; ++r) {
      int m = r + rowOffD;
      atomicAdd(&sOut[m], harr[r] * wq);  // ds_add_f32
    }
  }
  __syncthreads();
  if (tid < 16) {
    float qb = quantf(b_out[0]);
    out[blockIdx.x * 16 + tid] = quantf(sOut[tid] + qb);
  }
}

extern "C" void kernel_launch(void* const* d_in, const int* in_sizes, int n_in,
                              void* d_out, int out_size, void* d_ws, size_t ws_size,
                              hipStream_t stream) {
  const float* x    = (const float*)d_in[0];
  const float* wih  = (const float*)d_in[1];
  const float* whh  = (const float*)d_in[2];
  const float* bih  = (const float*)d_in[3];
  const float* bhh  = (const float*)d_in[4];
  const float* wout = (const float*)d_in[5];
  const float* bout = (const float*)d_in[6];
  float* out = (float*)d_out;

  dim3 grid(B_SZ / 16), block(128);
  hipLaunchKernelGGL(lstm_q88_wmma_kernel, grid, block, 0, stream,
                     x, wih, whh, bih, bhh, wout, bout, out);
}
